// MyViT_21715354649842
// MI455X (gfx1250) — matrix operand order repaired and checked by reading, not compiled, not run on hardware
//
#include <hip/hip_runtime.h>

// ---------------------------------------------------------------------------
// ViT patch embedding for MI455X (gfx1250, wave32, WMMA bf16 16x16x32)
//   out[n, 1+p, h] = sum_d patch[n,p,d] * W[h,d] + b[h] + pos[1+p, h]
//   out[n, 0,  h] = cls[h] + pos[0, h]
// GEMM: M = 128*196 = 25088, N = 768, K = 768
// ---------------------------------------------------------------------------

typedef __attribute__((ext_vector_type(16))) __bf16 v16bf;
typedef __attribute__((ext_vector_type(8)))  float  v8f;
typedef __attribute__((ext_vector_type(4)))  int    v4i;

#define NPATCH 196
#define DIM    768
#define SEQ    197
#define BATCH  128
#define TM     128
#define TN     128
#define PITCH  80          // bytes per LDS row: 32 bf16 (64B) + 16B pad
#define KSTEPS 24          // 768 / 32
#define ABYTES (TM * PITCH)            // 10240 B per buffer
#define BBASE  (2 * TM * PITCH)        // B buffers start at 20480

#if __has_builtin(__builtin_amdgcn_global_load_async_to_lds_b128)
#define USE_ASYNC_B 1
#endif

__device__ __forceinline__ unsigned int pk_bf16(float a, float b) {
#if __has_builtin(__builtin_amdgcn_cvt_pk_bf16_f32)
  auto r = __builtin_amdgcn_cvt_pk_bf16_f32(a, b);
  return __builtin_bit_cast(unsigned int, r);
#elif __has_builtin(__builtin_amdgcn_perm)
  // round-half-up to bf16, pack both halves with one v_perm_b32
  unsigned int ua = __float_as_uint(a) + 0x8000u;
  unsigned int ub = __float_as_uint(b) + 0x8000u;
  return __builtin_amdgcn_perm(ub, ua, 0x07060302u);  // {ub.hi16, ua.hi16}
#else
  unsigned int ua = __float_as_uint(a);
  unsigned int ub = __float_as_uint(b);
  ua = (ua + 0x7FFFu + ((ua >> 16) & 1u)) >> 16;          // RNE, low half
  ub = (ub + 0x7FFFu + ((ub >> 16) & 1u)) & 0xFFFF0000u;  // RNE, high half
  return ua | ub;
#endif
}

__device__ __forceinline__ void wait_async_lds() {
#if __has_builtin(__builtin_amdgcn_s_wait_asynccnt)
  __builtin_amdgcn_s_wait_asynccnt(0);
#elif defined(USE_ASYNC_B)
  asm volatile("s_wait_asynccnt 0x0" ::: "memory");
#endif
}

union Frag { uint4 q[2]; v16bf v; };

// ---------------------------------------------------------------------------
// W fp32 -> bf16 (row-major [DIM][DIM]) into workspace
// ---------------------------------------------------------------------------
__global__ __launch_bounds__(256)
void vit_convert_w(const float* __restrict__ W, unsigned short* __restrict__ wbf) {
  const int i = blockIdx.x * 256 + threadIdx.x;      // 589824/4 = 147456 threads
  const float4 f = ((const float4*)W)[i];
  uint2 u;
  u.x = pk_bf16(f.x, f.y);
  u.y = pk_bf16(f.z, f.w);
  ((uint2*)wbf)[i] = u;
}

// ---------------------------------------------------------------------------
// class-token rows: out[n, 0, :] = cls + pos[0, :]
// ---------------------------------------------------------------------------
__global__ __launch_bounds__(256)
void vit_cls_row(const float* __restrict__ cls, const float* __restrict__ pos,
                 float* __restrict__ out) {
  const int i = blockIdx.x * 256 + threadIdx.x;      // BATCH*DIM = 98304 threads
  const int n = i / DIM;
  const int j = i - n * DIM;
  out[(size_t)n * SEQ * DIM + j] = cls[j] + pos[j];
}

// ---------------------------------------------------------------------------
// Main fused patchify + GEMM (bf16 WMMA, f32 accum) + bias + pos-embed
// grid = (6 n-tiles, 196 m-tiles), block = 256 (8 waves, 4M x 2N)
// ---------------------------------------------------------------------------
__global__ __launch_bounds__(256)
void vit_patch_embed_gemm(const float* __restrict__ x,
                          const unsigned short* __restrict__ wbf,
                          const float* __restrict__ bias,
                          const float* __restrict__ pos,
                          float* __restrict__ out) {
  __shared__ __align__(16) unsigned char smem[4 * TM * PITCH];   // 40 KB

  const int tid  = threadIdx.x;
  const int lane = tid & 31;
  const int wav  = tid >> 5;
  const int wm   = (wav & 3) << 5;   // wave M offset: 0,32,64,96
  const int wn   = (wav >> 2) << 6;  // wave N offset: 0,64
  const int fl   = lane & 15;
  const int fh   = lane >> 4;

  const int mtile = blockIdx.y;      // 0..195
  const int ntile = blockIdx.x;      // 0..5

  // --- staging thread mapping: row sr (0..127), K-half sh (0/1) ---
  const int sr = tid >> 1;
  const int sh = tid & 1;

  // A source: patch sr of this m-tile; K chunk kt covers image rows
  //   ph0 = (kt&7)*2 (+sh) of channel c = kt>>3, cols px*16..px*16+15
  const int m   = mtile * TM + sr;
  const int nim = m / NPATCH;
  const int p   = m - nim * NPATCH;
  const int py  = p / 14;
  const int px  = p - py * 14;
  const float* xsrc =
      x + (size_t)nim * (3 * 224 * 224) + (py * 16 + sh) * 224 + px * 16;
  const unsigned short* wsrc =
      wbf + (size_t)(ntile * TN + sr) * DIM + sh * 16;

  // A tile: global fp32 -> regs -> bf16 -> LDS
  auto stageStoreA = [&](int buf, float4 f0, float4 f1, float4 f2, float4 f3) {
    uint4 u0, u1;
    u0.x = pk_bf16(f0.x, f0.y); u0.y = pk_bf16(f0.z, f0.w);
    u0.z = pk_bf16(f1.x, f1.y); u0.w = pk_bf16(f1.z, f1.w);
    u1.x = pk_bf16(f2.x, f2.y); u1.y = pk_bf16(f2.z, f2.w);
    u1.z = pk_bf16(f3.x, f3.y); u1.w = pk_bf16(f3.z, f3.w);
    uint4* ad = (uint4*)(smem + buf * ABYTES + sr * PITCH + sh * 32);
    ad[0] = u0; ad[1] = u1;
  };

#ifdef USE_ASYNC_B
  // B tile: already bf16 in workspace -> async DMA straight into LDS (ASYNCcnt)
  auto stageB = [&](int buf, int kt) {
    typedef __attribute__((address_space(1))) v4i g_v4i;
    typedef __attribute__((address_space(3))) v4i l_v4i;
    g_v4i* g = (g_v4i*)(wsrc + kt * 32);
    l_v4i* l = (l_v4i*)(smem + BBASE + buf * ABYTES + sr * PITCH + sh * 32);
    __builtin_amdgcn_global_load_async_to_lds_b128(g, l, 0, 0);
    __builtin_amdgcn_global_load_async_to_lds_b128(g + 1, l + 1, 0, 0);
  };
#else
  auto stageB = [&](int buf, int kt) {
    const uint4* bs = (const uint4*)(wsrc + kt * 32);
    uint4 b0 = bs[0], b1 = bs[1];
    uint4* bd = (uint4*)(smem + BBASE + buf * ABYTES + sr * PITCH + sh * 32);
    bd[0] = b0; bd[1] = b1;
  };
#endif

  v8f acc[2][4] = {};

  // ---- prologue: stage K chunk 0 into buffer 0 ----
  {
    stageB(0, 0);
    const float4* as = (const float4*)xsrc;          // kt=0: c=0, ph0=0
    stageStoreA(0, as[0], as[1], as[2], as[3]);
  }
  wait_async_lds();
  __syncthreads();

  for (int kt = 0; kt < KSTEPS; ++kt) {
    const int cur = kt & 1;

    // issue next K chunk's global traffic early (hide behind WMMA)
    float4 f0, f1, f2, f3;
    const bool more = (kt + 1) < KSTEPS;
    if (more) {
      const int kn = kt + 1;
      stageB(cur ^ 1, kn);                        // async -> LDS (or fallback)
      const float4* as =
          (const float4*)(xsrc + (kn >> 3) * 50176 + ((kn & 7) * 2) * 224);
      f0 = as[0]; f1 = as[1]; f2 = as[2]; f3 = as[3];
    }

    // ---- fragment loads from LDS (contiguous k = 16*h + s, same order in A and B) ----
    Frag afr[2], bfr[4];
#pragma unroll
    for (int fm = 0; fm < 2; ++fm) {
      const uint4* ap =
          (const uint4*)(smem + cur * ABYTES + (wm + fm * 16 + fl) * PITCH);
      afr[fm].q[0] = ap[2 * fh];       // K 0..7   (lanes 0-15) / 16..23 (16-31)
      afr[fm].q[1] = ap[2 * fh + 1];   // K 8..15  (lanes 0-15) / 24..31 (16-31)
    }
#pragma unroll
    for (int fn = 0; fn < 4; ++fn) {
      const uint4* bp = (const uint4*)(smem + BBASE + cur * ABYTES +
                                       (wn + fn * 16 + fl) * PITCH + fh * 32);
      bfr[fn].q[0] = bp[0];        // K 0..7  (+16 for lanes 16-31)
      bfr[fn].q[1] = bp[1];        // K 8..15 (+16 for lanes 16-31)
    }

#pragma unroll
    for (int fm = 0; fm < 2; ++fm)
#pragma unroll
      for (int fn = 0; fn < 4; ++fn)
        acc[fm][fn] = __builtin_amdgcn_wmma_f32_16x16x32_bf16(
            false, afr[fm].v, false, bfr[fn].v, (short)0, acc[fm][fn],
            false, false);

    if (more) stageStoreA(cur ^ 1, f0, f1, f2, f3);
    wait_async_lds();              // B tile landed in LDS before publishing
    __syncthreads();
  }

  // ---- epilogue: + bias + pos_embed, scatter to (n, 1+p, h) ----
#pragma unroll
  for (int fm = 0; fm < 2; ++fm) {
    const int mbase = mtile * TM + wm + fm * 16 + fh * 8;
#pragma unroll
    for (int fn = 0; fn < 4; ++fn) {
      const int ncol = ntile * TN + wn + fn * 16 + fl;
      const float bb = bias[ncol];
#pragma unroll
      for (int v = 0; v < 8; ++v) {
        const int mr = mbase + v;
        const int ni = mr / NPATCH;
        const int pp = mr - ni * NPATCH;
        const size_t o = ((size_t)ni * SEQ + 1 + pp) * DIM + ncol;
        out[o] = acc[fm][fn][v] + bb + pos[(1 + pp) * DIM + ncol];
      }
    }
  }
}

// ---------------------------------------------------------------------------
extern "C" void kernel_launch(void* const* d_in, const int* in_sizes, int n_in,
                              void* d_out, int out_size, void* d_ws, size_t ws_size,
                              hipStream_t stream) {
  const float* x   = (const float*)d_in[0];   // (128,3,224,224)
  const float* W   = (const float*)d_in[1];   // (768,768)
  const float* b   = (const float*)d_in[2];   // (768,)
  const float* cls = (const float*)d_in[3];   // (1,768)
  const float* pos = (const float*)d_in[4];   // (197,768)
  float* out = (float*)d_out;                 // (128,197,768)

  unsigned short* wbf = (unsigned short*)d_ws;  // 768*768*2 = 1.13 MB

  vit_convert_w<<<(DIM * DIM / 4) / 256, 256, 0, stream>>>(W, wbf);
  vit_cls_row<<<(BATCH * DIM) / 256, 256, 0, stream>>>(cls, pos, out);

  dim3 grid(DIM / TN, (BATCH * NPATCH) / TM);   // (6, 196)
  vit_patch_embed_gemm<<<grid, 256, 0, stream>>>(x, wbf, b, pos, out);
}
